// InfiniAttention_72911364817450
// MI455X (gfx1250) — compile-verified, matrix operand order checked by reading, not run on hardware
//
#include <hip/hip_runtime.h>
#include <hip/hip_bf16.h>
#include <stdint.h>

// ---------------------------------------------------------------------------
// InfiniAttention for MI455X (gfx1250): f16 WMMA everywhere, flash-style
// attention with fused Toeplitz relative-position term.
// Data movement: double-buffered global_load_async_to_lds_b128 in the GEMMs,
// Tensor Data Mover (tensor_load_to_lds + s_wait_tensorcnt) for K tiles.
// ---------------------------------------------------------------------------

typedef __attribute__((ext_vector_type(16))) _Float16     v16h;
typedef __attribute__((ext_vector_type(8)))  float        v8f;
typedef __attribute__((ext_vector_type(4)))  unsigned int u32x4;
typedef __attribute__((ext_vector_type(4)))  int          i32x4;
typedef __attribute__((ext_vector_type(8)))  int          i32x8;

#define DM    1024
#define SEQ   1024
#define BATCH 4
#define NH    16
#define HD    64
#define GM    4096   // rows of X / combined (B*S)

// LDS leading dims (halves). Row strides are multiples of 8 halves = 16B so
// b128 LDS writes stay naturally aligned.
#define AL_LD 40
#define WT_LD 40
#define KT_LD 72     // 64 data halves + 8 pad halves (TDM pad: 4 dwords / 32)
#define VT_LD 40
#define ET_LD 40

__device__ __forceinline__ v8f zero8() {
  v8f z;
#pragma unroll
  for (int i = 0; i < 8; ++i) z[i] = 0.0f;
  return z;
}

// A/B^T fragment loader for V_WMMA_F32_16X16X32_F16 per CDNA5 ISA 7.12.2:
// lane l: row m = l&15; halves i=0..7 -> K = hi8 + i, i=8..15 -> K = 16+hi8+(i-8)
// where hi8 = 8 for lanes 16..31. Works for row-major A, and for B when given
// B^T rows (i.e. column-major B).
__device__ __forceinline__ v16h load_frag16(const _Float16* __restrict__ src, int ld) {
  const int l   = threadIdx.x & 31;
  const int m   = l & 15;
  const int hi8 = (l >> 4) << 3;
  v16h f;
#pragma unroll
  for (int i = 0; i < 16; ++i) {
    const int k = ((i >> 3) << 4) + hi8 + (i & 7);
    f[i] = src[m * ld + k];
  }
  return f;
}

__device__ __forceinline__ v8f wmma16(v16h a, v16h b, v8f c) {
  // D = A(16x32 f16) * B(32x16 f16) + C(16x16 f32)
  return __builtin_amdgcn_wmma_f32_16x16x32_f16(
      /*neg_a=*/false, a, /*neg_b=*/false, b,
      /*c_mod=*/(short)0, c, /*reuse_a=*/false, /*reuse_b=*/false);
}

// CDNA5 async copy: 16B global -> LDS per lane, tracked by ASYNCcnt.
__device__ __forceinline__ void async_copy_b128(uint32_t lds_off, const void* gptr) {
  asm volatile("global_load_async_to_lds_b128 %0, %1, off"
               :: "v"(lds_off), "v"((uint64_t)(uintptr_t)gptr)
               : "memory");
}
__device__ __forceinline__ void wait_async() {
  asm volatile("s_wait_asynccnt 0" ::: "memory");
}
__device__ __forceinline__ void wait_ds() {
  asm volatile("s_wait_dscnt 0" ::: "memory");
}

// ---------------------------------------------------------------------------
// Tensor Data Mover: DMA a 2-D f16 tile (tile_rows x row_halves) into LDS,
// with hardware padding of 4 dwords after every 32 dwords so each 64-half
// row lands on a 72-half (144B) LDS stride. D# bitfields per ISA 8.3/8.4.
// ---------------------------------------------------------------------------
__device__ __forceinline__ void tdm_load_tile_f16(uint32_t lds_byte_off,
                                                  const void* gaddr,
                                                  uint32_t tile_rows,
                                                  uint32_t row_halves,
                                                  uint32_t tensor_rows) {
  const uint64_t ga = (uint64_t)(uintptr_t)gaddr;
  u32x4 g0 = {0u, 0u, 0u, 0u};
  g0[0] = 1u;                                   // count=1, user descriptor
  g0[1] = lds_byte_off;                         // lds_addr
  g0[2] = (uint32_t)(ga & 0xffffffffu);         // global_addr[31:0]
  g0[3] = (uint32_t)(ga >> 32) | (2u << 30);    // global_addr[56:32] | type=2

  i32x8 g1 = {0, 0, 0, 0, 0, 0, 0, 0};
  // workgroup_mask=0, data_size=1 (2B), pad_enable=1,
  // pad_interval=4 (32 dwords), pad_amount=3 (4 dwords)
  g1[0] = (int)((1u << 16) | (1u << 20) | (4u << 22) | (3u << 25));
  g1[1] = (int)((row_halves & 0xffffu) << 16);                       // dim0[15:0]
  g1[2] = (int)(((row_halves >> 16) & 0xffffu) |
                ((tensor_rows & 0xffffu) << 16));                    // dim0[31:16]|dim1[15:0]
  g1[3] = (int)(((tensor_rows >> 16) & 0xffffu) |
                ((row_halves & 0xffffu) << 16));                     // dim1[31:16]|tile_dim0
  g1[4] = (int)(tile_rows & 0xffffu);                                // tile_dim1, tile_dim2=0
  g1[5] = (int)row_halves;                                           // dim0_stride[31:0]
  g1[6] = 0;                                                         // stride hi / dim1_stride
  g1[7] = 0;

  const i32x4 gz = {0, 0, 0, 0};
#if __clang_major__ >= 23
  const i32x8 gz8 = {0, 0, 0, 0, 0, 0, 0, 0};
  __builtin_amdgcn_tensor_load_to_lds(g0, g1, gz, gz, gz8, 0);
#else
  __builtin_amdgcn_tensor_load_to_lds(g0, g1, gz, gz, 0);
#endif
}
__device__ __forceinline__ void wait_tensor() {
  __builtin_amdgcn_s_wait_tensorcnt((short)0);
}

// ---------------------------------------------------------------------------
// K1: f32 -> f16 conversion
// ---------------------------------------------------------------------------
__global__ void cvt_f32_to_f16(const float* __restrict__ in,
                               _Float16* __restrict__ out, int n) {
  int i = blockIdx.x * blockDim.x + threadIdx.x;
  if (i < n) out[i] = (_Float16)in[i];
}

// ---------------------------------------------------------------------------
// K2/K5: C[4096,1024] = A[4096,1024] @ W[1024,1024] + bias
// Block tile 128x64, 8 waves, each wave one 16x64 strip (4 WMMA frag cols).
// Double-buffered LDS: async copies for chunk k+1 overlap WMMA on chunk k.
// MODE 0: write f16 into [b,h,s,d] layout (QKV). MODE 1: write f32 row-major.
// ---------------------------------------------------------------------------
template <int MODE>
__global__ __launch_bounds__(256)
void gemm4096_wmma(const _Float16* __restrict__ A,
                   const _Float16* __restrict__ W,
                   const float* __restrict__ bias,
                   void* __restrict__ out) {
  __shared__ _Float16 Al[2][128][AL_LD];
  __shared__ _Float16 Wt[2][64][WT_LD];   // W tile transposed: Wt[n][k]

  const int bm0 = blockIdx.x * 128;
  const int bn0 = blockIdx.y * 64;
  const int tid = threadIdx.x;
  const int w   = tid >> 5;
  const int l   = tid & 31;
  const int n   = l & 15;
  const int hi8 = (l >> 4) << 3;

  auto stage = [&](int kk, int buf) {
    // A tile 128x32 halves via async-to-LDS (512 x b128, 2 per thread)
#pragma unroll
    for (int it = 0; it < 2; ++it) {
      const int e   = tid + it * 256;
      const int row = e >> 2;
      const int c8  = (e & 3) * 8;
      async_copy_b128((uint32_t)(uintptr_t)&Al[buf][row][c8],
                      &A[(size_t)(bm0 + row) * 1024 + kk + c8]);
    }
    // W tile 32x64, stored transposed so the B fragment reads contiguously
    const int row = tid >> 3;          // k within tile: 0..31
    const int c8  = (tid & 7) * 8;     // n base: 0..56
    const _Float16* src = &W[(size_t)(kk + row) * 1024 + bn0 + c8];
#pragma unroll
    for (int j = 0; j < 8; ++j) Wt[buf][c8 + j][row] = src[j];
  };

  v8f acc[4];
#pragma unroll
  for (int i = 0; i < 4; ++i) acc[i] = zero8();

  stage(0, 0);
  int buf = 0;
  for (int kk = 0; kk < 1024; kk += 32, buf ^= 1) {
    wait_async();          // chunk kk landed (issued last iteration)
    __syncthreads();       // publish LDS writes across waves
    if (kk + 32 < 1024) stage(kk + 32, buf ^ 1);   // overlap DMA with WMMA

    const v16h a = load_frag16(&Al[buf][w * 16][0], AL_LD);
#pragma unroll
    for (int nb = 0; nb < 4; ++nb) {
      const v16h bf = load_frag16(&Wt[buf][nb * 16][0], WT_LD);
      acc[nb] = wmma16(a, bf, acc[nb]);
    }
  }

#pragma unroll
  for (int nb = 0; nb < 4; ++nb) {
#pragma unroll
    for (int r = 0; r < 8; ++r) {
      const int row = bm0 + w * 16 + r + hi8;
      const int col = bn0 + nb * 16 + n;
      const float v = acc[nb][r] + bias[col];
      if (MODE == 0) {
        const int b = row >> 10, sr = row & 1023, hh = col >> 6, dd = col & 63;
        ((_Float16*)out)[(((size_t)(b * NH + hh)) * SEQ + sr) * HD + dd] = (_Float16)v;
      } else {
        ((float*)out)[(size_t)row * 1024 + col] = v;
      }
    }
  }
}

// ---------------------------------------------------------------------------
// K3: flash attention with fused relative-position scores.
// Block = (b, h, 128-row q block); 8 waves, each wave a 16-row q tile.
// K chunks arrive via TDM (tensor_load_to_lds, padded rows); V chunks are
// transposed into LDS for the exp(S)@V B-fragment. Online softmax per row.
// Writes local attention output (pre-gating) to Ch[b,s,h*64+d] f16.
// ---------------------------------------------------------------------------
__global__ __launch_bounds__(256)
void attn_flash_wmma(const _Float16* __restrict__ Qh,
                     const _Float16* __restrict__ Kh,
                     const _Float16* __restrict__ Vh,
                     const _Float16* __restrict__ relh,  // [2049,64]
                     const float* __restrict__ mask,     // [B,S]
                     _Float16* __restrict__ Ch) {        // [B,S,DM]
  __shared__ _Float16 Kt[32][KT_LD];       // K chunk row-major [k][d] (TDM-filled)
  __shared__ _Float16 Vt[64][VT_LD];       // V chunk transposed [d][k]
  __shared__ _Float16 Et[8][16][ET_LD];    // per-wave exp(S) tile (A-frag src)
  __shared__ float    Pg[8][32][16];       // per-wave rel-score gather buffer

  const int qblk = blockIdx.x & 7;         // S/128 = 8
  const int bh   = blockIdx.x >> 3;
  const int h    = bh & (NH - 1);
  const int bb   = bh >> 4;
  const int tid  = threadIdx.x;
  const int w    = tid >> 5;
  const int l    = tid & 31;
  const int n    = l & 15;
  const int hi8  = (l >> 4) << 3;
  const int q0   = qblk * 128 + w * 16;    // absolute q of this wave's tile

  const size_t baseBH = ((size_t)(bb * NH + h)) * SEQ * HD;
  const _Float16* Qp = Qh + baseBH + (size_t)q0 * HD;
  const v16h aq0 = load_frag16(Qp + 0,  HD);   // d = 0..31
  const v16h aq1 = load_frag16(Qp + 32, HD);   // d = 32..63

  v8f accd[4];
#pragma unroll
  for (int i = 0; i < 4; ++i) accd[i] = zero8();
  float mrow[8], lrow[8];
#pragma unroll
  for (int r = 0; r < 8; ++r) { mrow[r] = -1e30f; lrow[r] = 0.0f; }

  for (int k0 = 0; k0 < SEQ; k0 += 32) {
    // --- stage K chunk via Tensor Data Mover (one wave issues, EXEC-ignored)
    if (w == 0) {
      tdm_load_tile_f16((uint32_t)(uintptr_t)&Kt[0][0],
                        &Kh[baseBH + (size_t)k0 * HD],
                        /*tile_rows=*/32, /*row_halves=*/HD,
                        /*tensor_rows=*/(uint32_t)(SEQ - k0));
    }
    // --- stage V chunk transposed (regular loads + LDS scatter) ---
    {
      const int row = tid >> 3;         // k within chunk 0..31
      const int c8  = (tid & 7) * 8;    // d base 0..56
      const _Float16* vs = &Vh[baseBH + (size_t)(k0 + row) * HD + c8];
#pragma unroll
      for (int j = 0; j < 8; ++j) Vt[c8 + j][row] = vs[j];
    }
    if (w == 0) wait_tensor();
    __syncthreads();

    v16h bv[4];
#pragma unroll
    for (int nb = 0; nb < 4; ++nb) bv[nb] = load_frag16(&Vt[nb * 16][0], VT_LD);

    float s2[2][8];
#pragma unroll
    for (int sub = 0; sub < 2; ++sub) {
      const int ks0 = k0 + sub * 16;
      const v16h bk0 = load_frag16(&Kt[sub * 16][0],  KT_LD);
      const v16h bk1 = load_frag16(&Kt[sub * 16][32], KT_LD);

      // content term: Q @ K^T over d=64
      v8f s = zero8();
      s = wmma16(aq0, bk0, s);
      s = wmma16(aq1, bk1, s);

      // rel term: P[i,dk] = <rel_emb[jb+i], K[ks0+dk]>, i in [0,31]
      const int jb = q0 - ks0 + (1024 - 15);   // always in [1, 2017]
      v8f p0 = zero8(), p1 = zero8();
      p0 = wmma16(load_frag16(relh + (size_t)jb * HD,             HD), bk0, p0);
      p0 = wmma16(load_frag16(relh + (size_t)jb * HD + 32,        HD), bk1, p0);
      p1 = wmma16(load_frag16(relh + (size_t)(jb + 16) * HD,      HD), bk0, p1);
      p1 = wmma16(load_frag16(relh + (size_t)(jb + 16) * HD + 32, HD), bk1, p1);
#pragma unroll
      for (int r = 0; r < 8; ++r) {
        Pg[w][r + hi8][n]      = p0[r];
        Pg[w][16 + r + hi8][n] = p1[r];
      }
      wait_ds();   // wave-local LDS RAW

      const float mval = mask[bb * SEQ + ks0 + n] * -1e9f;
#pragma unroll
      for (int r = 0; r < 8; ++r) {
        const int m = r + hi8;
        s2[sub][r] = (s[r] + Pg[w][m - n + 15][n]) * 0.125f + mval;
      }
    }

    // --- online softmax (rows live across 16 lanes; 8 rows per lane) ---
#pragma unroll
    for (int r = 0; r < 8; ++r) {
      float rm = fmaxf(s2[0][r], s2[1][r]);
#pragma unroll
      for (int off = 1; off < 16; off <<= 1) rm = fmaxf(rm, __shfl_xor(rm, off, 32));
      const float mnew = fmaxf(mrow[r], rm);
      const float ef   = __expf(mrow[r] - mnew);
      const float e0   = __expf(s2[0][r] - mnew);
      const float e1   = __expf(s2[1][r] - mnew);
      float rs = e0 + e1;
#pragma unroll
      for (int off = 1; off < 16; off <<= 1) rs += __shfl_xor(rs, off, 32);
      lrow[r] = lrow[r] * ef + rs;
      mrow[r] = mnew;
      const int m = r + hi8;
      Et[w][m][n]      = (_Float16)e0;
      Et[w][m][16 + n] = (_Float16)e1;
#pragma unroll
      for (int nb = 0; nb < 4; ++nb) accd[nb][r] *= ef;
    }
    wait_ds();

    // acc += exp(S) @ V  (A-frag from Et, B-frag from transposed V)
    const v16h ae = load_frag16(&Et[w][0][0], ET_LD);
#pragma unroll
    for (int nb = 0; nb < 4; ++nb) accd[nb] = wmma16(ae, bv[nb], accd[nb]);

    __syncthreads();   // protect Kt/Vt before next chunk
  }

  // local attention output -> Ch[b, q, h*64+d]
#pragma unroll
  for (int nb = 0; nb < 4; ++nb) {
#pragma unroll
    for (int r = 0; r < 8; ++r) {
      const int m = r + hi8;
      const float v = accd[nb][r] / lrow[r];
      Ch[((size_t)bb * SEQ + q0 + m) * DM + h * HD + nb * 16 + n] = (_Float16)v;
    }
  }
}

// ---------------------------------------------------------------------------
// K4: linear-attention global read (elu(q)+1 vs M, z) + sigmoid gating.
// Tiny FLOP count (~0.5 GF) -> plain VALU, one thread per (b,h,q,d).
// ---------------------------------------------------------------------------
__global__ __launch_bounds__(256)
void global_memory_gate(const _Float16* __restrict__ Qh,
                        const float* __restrict__ Mrd,  // [H,D,D]
                        const float* __restrict__ z,    // [H,D]
                        const float* __restrict__ galpha,
                        _Float16* __restrict__ Ch) {
  const int t = blockIdx.x * blockDim.x + threadIdx.x;
  const int d = t & 63;
  const int q = (t >> 6) & 1023;
  const int h = (t >> 16) & 15;
  const int b = t >> 20;
  const _Float16* qrow = Qh + (((size_t)(b * NH + h)) * SEQ + q) * HD;
  float num = 0.0f, den = 1e-6f;
#pragma unroll 8
  for (int dd = 0; dd < 64; ++dd) {
    const float x = (float)qrow[dd];
    const float s = x > 0.0f ? x + 1.0f : __expf(x);   // elu(x)+1
    den += s * z[h * 64 + dd];
    num += s * Mrd[((size_t)h * 64 + dd) * 64 + d];
  }
  const float glob = num / den;
  const float gate = 1.0f / (1.0f + __expf(-galpha[0]));
  const size_t ci = ((size_t)b * SEQ + q) * DM + h * HD + d;
  const float local = (float)Ch[ci];
  Ch[ci] = (_Float16)(gate * local + (1.0f - gate) * glob);
}

// ---------------------------------------------------------------------------
// Host-side orchestration
// ---------------------------------------------------------------------------
extern "C" void kernel_launch(void* const* d_in, const int* in_sizes, int n_in,
                              void* d_out, int out_size, void* d_ws, size_t ws_size,
                              hipStream_t stream) {
  (void)in_sizes; (void)n_in; (void)out_size; (void)ws_size;
  const float* hidden = (const float*)d_in[0];
  const float* mask   = (const float*)d_in[1];
  const float* Wq = (const float*)d_in[2];  const float* bq = (const float*)d_in[3];
  const float* Wk = (const float*)d_in[4];  const float* bk = (const float*)d_in[5];
  const float* Wv = (const float*)d_in[6];  const float* bv = (const float*)d_in[7];
  const float* Wo = (const float*)d_in[8];  const float* bo = (const float*)d_in[9];
  const float* rel = (const float*)d_in[10];
  const float* M   = (const float*)d_in[11];
  const float* z   = (const float*)d_in[12];
  const float* ga  = (const float*)d_in[13];

  // workspace carve-up (f16), ~48 MB total
  _Float16* p   = (_Float16*)d_ws;
  _Float16* Xh  = p; p += (size_t)GM * DM;
  _Float16* Wqh = p; p += (size_t)DM * DM;
  _Float16* Wkh = p; p += (size_t)DM * DM;
  _Float16* Wvh = p; p += (size_t)DM * DM;
  _Float16* Woh = p; p += (size_t)DM * DM;
  _Float16* relh = p; p += (size_t)2049 * HD;
  _Float16* Qh  = p; p += (size_t)GM * DM;
  _Float16* Kh  = p; p += (size_t)GM * DM;
  _Float16* Vh  = p; p += (size_t)GM * DM;
  _Float16* Ch  = p; p += (size_t)GM * DM;

  auto cvt = [&](const float* src, _Float16* dst, int n) {
    cvt_f32_to_f16<<<(n + 255) / 256, 256, 0, stream>>>(src, dst, n);
  };
  cvt(hidden, Xh, GM * DM);
  cvt(Wq, Wqh, DM * DM);
  cvt(Wk, Wkh, DM * DM);
  cvt(Wv, Wvh, DM * DM);
  cvt(Wo, Woh, DM * DM);
  cvt(rel, relh, 2049 * HD);

  dim3 gg(GM / 128, DM / 64);   // 32 x 16
  gemm4096_wmma<0><<<gg, 256, 0, stream>>>(Xh, Wqh, bq, (void*)Qh);
  gemm4096_wmma<0><<<gg, 256, 0, stream>>>(Xh, Wkh, bk, (void*)Kh);
  gemm4096_wmma<0><<<gg, 256, 0, stream>>>(Xh, Wvh, bv, (void*)Vh);

  attn_flash_wmma<<<BATCH * NH * (SEQ / 128), 256, 0, stream>>>(
      Qh, Kh, Vh, relh, mask, Ch);

  global_memory_gate<<<(BATCH * NH * SEQ * HD) / 256, 256, 0, stream>>>(
      Qh, M, z, ga, Ch);

  gemm4096_wmma<1><<<gg, 256, 0, stream>>>(Ch, Woh, bo, d_out);
}
